// AttentionBlock_31808527794843
// MI455X (gfx1250) — matrix operand order repaired and checked by reading, not compiled, not run on hardware
//
#include <hip/hip_runtime.h>

// ---------------------------------------------------------------------------
// MI455X / gfx1250 transformer block: rmsnorm -> QKV(+RoPE) -> flash-attn
// (alibi softmax) -> out-proj(+residual) -> rmsnorm -> MLP (gelu) -> residual.
// GEMMs: v_wmma_f32_16x16x32_bf16 (wave32), 32x64 tile per wave, block-shared
// A tile staged to LDS via async-to-LDS DMA (ASYNCcnt), weights bf16 in L2.
// ---------------------------------------------------------------------------

typedef __bf16 bf16;
typedef __attribute__((ext_vector_type(16))) __bf16 v16bf;
typedef __attribute__((ext_vector_type(8)))  float  v8f;

#define Bsz 2
#define Tsz 2048
#define Csz 1024
#define Hsz 16
#define Dsz 64
#define Fsz 4096
#define BT  (Bsz * Tsz)
#define HD  (Hsz * Dsz)

#if __has_builtin(__builtin_amdgcn_global_load_async_to_lds_b128)
#define HAVE_ASYNC_LDS 1
// Builtin signature (from hipcc diagnostic): first param is
// 'int __vector(4) __device__ *' i.e. AS(1) pointer to a 16B int vector.
typedef int vint4 __attribute__((vector_size(16)));
typedef __attribute__((address_space(1))) vint4 as1_vint4;
typedef __attribute__((address_space(3))) vint4 as3_vint4;
#endif

union FragU { uint4 u[2]; v16bf v; };

static __device__ __forceinline__ v8f zero8() {
  v8f z = {0.f, 0.f, 0.f, 0.f, 0.f, 0.f, 0.f, 0.f};
  return z;
}

static __device__ __forceinline__ v8f wmma_bf16(v16bf a, v16bf b, v8f c) {
  // D = A(16x32) * B(32x16) + C(16x16 f32)
  return __builtin_amdgcn_wmma_f32_16x16x32_bf16(
      /*neg_a=*/false, a, /*neg_b=*/false, b,
      /*c_mod=*/(short)0, c, /*reuse_a=*/false, /*reuse_b=*/false);
}

// A-fragment: 16(M) x 32(K), row-major source, `tile` points at (m0,k0),
// row stride lda (elements).  Lane l holds row M=l; hi selects K half:
// VGPR0-7 = K[hi*16 .. hi*16+15].
static __device__ __forceinline__ v16bf load_a_frag(const bf16* tile, int lda,
                                                    int l, int hi) {
  FragU f;
  const bf16* p = tile + (size_t)l * lda + hi * 16;
  f.u[0] = *(const uint4*)(p);
  f.u[1] = *(const uint4*)(p + 8);
  return f.v;
}

// B-fragment: B[k][n] = src[n][k] (i.e. A*B^T GEMM), src row-major with row
// stride lds, `tile` points at src[n0][k0].  Lane (l,hi): N=l, VGPR i holds
// K = hi*16 + 2i, 2i+1  ->  16 consecutive K elements = two b128 loads.
static __device__ __forceinline__ v16bf load_bT_frag(const bf16* tile, int lds,
                                                     int l, int hi) {
  FragU f;
  const bf16* p = tile + (size_t)l * lds + hi * 16;
  f.u[0] = *(const uint4*)(p);
  f.u[1] = *(const uint4*)(p + 8);
  return f.v;
}

// ---------------------------------------------------------------------------
// Block-cooperative staging of a 32(M) x 32(K) bf16 A tile into LDS.
// 2 KB per tile = 128 x 16B chunks; wave 0 moves it (4 b128 per lane).
// Async DMA path (ASYNCcnt) when the toolchain exposes the builtin.
// ---------------------------------------------------------------------------
static __device__ __forceinline__ void wait_async0() {
#if __has_builtin(__builtin_amdgcn_s_wait_asynccnt)
  __builtin_amdgcn_s_wait_asynccnt(0);
#else
  asm volatile("s_wait_asynccnt 0x0" ::: "memory");
#endif
}

static __device__ __forceinline__ void stage_a_tile(const bf16* gbase, int K,
                                                    bf16* lbase, int lane) {
#pragma unroll
  for (int t = 0; t < 4; ++t) {
    const int c = t * 32 + lane;          // 16B chunk id in [0,128)
    const int row = c >> 2;               // 4 chunks per 64B row
    const int cb = (c & 3) * 16;          // byte offset within row
    const char* g = (const char*)gbase + (size_t)row * K * 2 + cb;
    char* lp = (char*)lbase + c * 16;
#ifdef HAVE_ASYNC_LDS
    __builtin_amdgcn_global_load_async_to_lds_b128(
        (as1_vint4*)(unsigned long long)g,
        (as3_vint4*)(unsigned long long)lp, 0, 0);
#else
    *(uint4*)lp = *(const uint4*)g;
#endif
  }
}

static __device__ __forceinline__ void stage_a_wait() {
#ifdef HAVE_ASYNC_LDS
  wait_async0();
#endif
}

// ---------------------------------------------------------------------------
// f32 -> bf16 elementwise (one-time weight conversion; thereafter L2-resident)
// ---------------------------------------------------------------------------
__global__ void f2bf_kernel(const float* __restrict__ in, bf16* __restrict__ out,
                            int n) {
  for (int i = blockIdx.x * blockDim.x + threadIdx.x; i < n;
       i += gridDim.x * blockDim.x)
    out[i] = (bf16)in[i];
}

// ---------------------------------------------------------------------------
// RMSNorm: one row of Cdim per block (256 threads), f32 in, bf16 out
// ---------------------------------------------------------------------------
__global__ void rmsnorm_kernel(const float* __restrict__ x,
                               const float* __restrict__ g,
                               bf16* __restrict__ y, int Cdim) {
  __shared__ float red[256];
  const int row = blockIdx.x;
  const float* xr = x + (size_t)row * Cdim;
  float s = 0.f;
  for (int c = threadIdx.x; c < Cdim; c += 256) {
    float v = xr[c];
    s += v * v;
  }
  red[threadIdx.x] = s;
  __syncthreads();
  for (int off = 128; off > 0; off >>= 1) {
    if ((int)threadIdx.x < off) red[threadIdx.x] += red[threadIdx.x + off];
    __syncthreads();
  }
  const float inv = rsqrtf(red[0] / (float)Cdim + 1e-8f);
  bf16* yr = y + (size_t)row * Cdim;
  for (int c = threadIdx.x; c < Cdim; c += 256)
    yr[c] = (bf16)(xr[c] * inv * g[c]);
}

// ---------------------------------------------------------------------------
// QKV GEMM with fused RoPE.  Each wave: 16 tokens x one head-dim (N=64) of
// one of {q,k,v}.  rotate_half pairs (d, d+-32) sit in the same lane of
// acc[j] and acc[j^2] -> rope needs no cross-lane traffic.
// v is stored transposed [bh][D][T] so attn's P*V B-frags are contiguous.
// ---------------------------------------------------------------------------
__global__ void qkv_rope_kernel(const bf16* __restrict__ yn,
                                const bf16* __restrict__ wqkv,
                                const float* __restrict__ rot,
                                bf16* __restrict__ qb, bf16* __restrict__ kb,
                                bf16* __restrict__ vTb) {
  const int lane = threadIdx.x & 31, l = lane & 15, hi = lane >> 4;
  const int w = blockIdx.x * 8 + (threadIdx.x >> 5);
  const int mtile = w / 48;           // 256 token tiles
  const int etile = w % 48;           // 48 = 3 * 16 heads
  const int which = etile >> 4;       // 0=q 1=k 2=v
  const int h = etile & 15;
  const int m0 = mtile * 16;
  const int e0 = which * HD + h * Dsz;

  v8f acc[4] = {zero8(), zero8(), zero8(), zero8()};
  for (int k0 = 0; k0 < Csz; k0 += 32) {
    v16bf a = load_a_frag(yn + (size_t)m0 * Csz + k0, Csz, l, hi);
    __builtin_prefetch(yn + (size_t)m0 * Csz + k0 + 64, 0, 2);
#pragma unroll
    for (int j = 0; j < 4; ++j)
      acc[j] = wmma_bf16(
          a, load_bT_frag(wqkv + (size_t)(e0 + 16 * j) * Csz + k0, Csz, l, hi),
          acc[j]);
  }

#pragma unroll
  for (int r = 0; r < 8; ++r) {
    const int row = m0 + hi * 8 + r;      // flat token index in [0, B*T)
    const int b = row >> 11;              // / Tsz
    const int t = row & (Tsz - 1);
    const size_t bh = (size_t)(b * Hsz + h);
    if (which == 2) {
#pragma unroll
      for (int j = 0; j < 4; ++j) {
        const int d = 16 * j + l;
        vTb[(bh * Dsz + d) * Tsz + t] = (bf16)acc[j][r];
      }
    } else {
      bf16* dst = (which == 0) ? qb : kb;
#pragma unroll
      for (int j = 0; j < 4; ++j) {
        const int d = 16 * j + l;
        const float pos = rot[(size_t)t * Dsz + d];
        const float other = (j < 2) ? -acc[j + 2][r] : acc[j - 2][r];
        const float val = acc[j][r] * cosf(pos) + other * sinf(pos);
        dst[(bh * Tsz + t) * Dsz + d] = (bf16)val;
      }
    }
  }
}

// ---------------------------------------------------------------------------
// Flash attention with alibi.  One wave = one (b,h,16-row q-tile); streams
// 32-key chunks.  Online softmax state (m,l) replicated across each 16-lane
// half via __shfl_xor(width=16) since C-layout rows live per half-wave.
// P is relaid out C-layout -> A-layout through a per-wave LDS tile.
// ---------------------------------------------------------------------------
__global__ void attn_kernel(const bf16* __restrict__ q,
                            const bf16* __restrict__ k,
                            const bf16* __restrict__ vT,
                            const float* __restrict__ alibi,
                            bf16* __restrict__ attY) {
  __shared__ bf16 sP[4][16 * 32];
  const int lane = threadIdx.x & 31, l = lane & 15, hi = lane >> 4;
  const int wv = threadIdx.x >> 5;
  const int qt = blockIdx.x * 4 + wv;     // 4096 q-tiles
  const int bh = qt >> 7;                 // 128 q-tiles per (b,h)
  const int q0 = (qt & 127) << 4;
  const int h = bh & (Hsz - 1);
  const int b = bh >> 4;

  const bf16* qbase = q + ((size_t)bh * Tsz + q0) * Dsz;
  const bf16* kbase = k + (size_t)bh * Tsz * Dsz;
  const bf16* vbase = vT + (size_t)bh * Dsz * Tsz;
  const float* albase = alibi + ((size_t)h * Tsz + q0) * Tsz;

  const v16bf aq0 = load_a_frag(qbase, Dsz, l, hi);
  const v16bf aq1 = load_a_frag(qbase + 32, Dsz, l, hi);

  float mrow[8], lrow[8];
  v8f O[4] = {zero8(), zero8(), zero8(), zero8()};
#pragma unroll
  for (int r = 0; r < 8; ++r) { mrow[r] = -1e30f; lrow[r] = 0.f; }

  for (int kc = 0; kc < Tsz; kc += 32) {
    v8f S0 = zero8(), S1 = zero8();
    S0 = wmma_bf16(aq0, load_bT_frag(kbase + (size_t)kc * Dsz, Dsz, l, hi), S0);
    S0 = wmma_bf16(aq1, load_bT_frag(kbase + (size_t)kc * Dsz + 32, Dsz, l, hi), S0);
    S1 = wmma_bf16(aq0, load_bT_frag(kbase + (size_t)(kc + 16) * Dsz, Dsz, l, hi), S1);
    S1 = wmma_bf16(aq1, load_bT_frag(kbase + (size_t)(kc + 16) * Dsz + 32, Dsz, l, hi), S1);

#pragma unroll
    for (int r = 0; r < 8; ++r) {
      const size_t arow = (size_t)(hi * 8 + r) * Tsz + kc;
      S0[r] = S0[r] * 0.125f + albase[arow + l];
      S1[r] = S1[r] * 0.125f + albase[arow + 16 + l];

      float tmax = fmaxf(S0[r], S1[r]);
      for (int off = 1; off < 16; off <<= 1)
        tmax = fmaxf(tmax, __shfl_xor(tmax, off, 16));
      const float mn = fmaxf(mrow[r], tmax);
      const float sf = __expf(mrow[r] - mn);
      const float p0 = __expf(S0[r] - mn);
      const float p1 = __expf(S1[r] - mn);
      float rs = p0 + p1;
      for (int off = 1; off < 16; off <<= 1) rs += __shfl_xor(rs, off, 16);
      lrow[r] = lrow[r] * sf + rs;
      mrow[r] = mn;
#pragma unroll
      for (int j = 0; j < 4; ++j) O[j][r] *= sf;
      sP[wv][(hi * 8 + r) * 32 + l] = (bf16)p0;
      sP[wv][(hi * 8 + r) * 32 + 16 + l] = (bf16)p1;
    }

    const v16bf pa = load_a_frag((const bf16*)sP[wv], 32, l, hi);
#pragma unroll
    for (int j = 0; j < 4; ++j)
      O[j] = wmma_bf16(
          pa, load_bT_frag(vbase + (size_t)(16 * j) * Tsz + kc, Tsz, l, hi),
          O[j]);
  }

#pragma unroll
  for (int r = 0; r < 8; ++r) {
    const float inv = 1.f / lrow[r];
    const int row = b * Tsz + q0 + hi * 8 + r;
#pragma unroll
    for (int j = 0; j < 4; ++j)
      attY[(size_t)row * HD + h * Dsz + 16 * j + l] = (bf16)(O[j][r] * inv);
  }
}

// ---------------------------------------------------------------------------
// Generic A(bf16,[M,K]) x W(bf16,[N,K])^T GEMM.
// Block = 8 waves sharing one 32-row A tile (double-buffered in LDS, staged
// by wave 0 via async-to-LDS DMA); each wave owns a 32(M) x 64(N) tile:
// one B-fragment feeds two WMMAs (2x arithmetic intensity on W).
// MODE 0: out_f32 = resid + acc                    (attention out-proj)
// MODE 1: out_bf16 = gelu(acc + bias)              (MLP in)
// MODE 2: out_f32 = resid + bias + acc             (MLP out)
// ---------------------------------------------------------------------------
template <int MODE>
__global__ __launch_bounds__(256) void gemm32x512_kernel(
    const bf16* __restrict__ A, const bf16* __restrict__ W,
    const float* __restrict__ bias, const float* __restrict__ resid,
    float* __restrict__ outf, bf16* __restrict__ outb, int M, int N, int K) {
  __shared__ bf16 sA[2][32 * 32];       // double-buffered 2 KB A tiles
  const int lane = threadIdx.x & 31, l = lane & 15, hi = lane >> 4;
  const int wv = threadIdx.x >> 5;
  const int ntiles = N >> 9;            // 512-wide block tiles
  const int m0 = (blockIdx.x / ntiles) * 32;
  const int n0 = (blockIdx.x % ntiles) * 512 + wv * 64;
  const bf16* Abase = A + (size_t)m0 * K;

  if (wv == 0) {
    stage_a_tile(Abase, K, sA[0], lane);
    stage_a_wait();
  }
  __syncthreads();

  v8f acc[8] = {zero8(), zero8(), zero8(), zero8(),
                zero8(), zero8(), zero8(), zero8()};
  const int nk = K >> 5;
  for (int i = 0; i < nk; ++i) {
    const int k0 = i << 5;
    const int cur = i & 1;
    if (wv == 0 && i + 1 < nk)
      stage_a_tile(Abase + k0 + 32, K, sA[cur ^ 1], lane);

    const v16bf a0 = load_a_frag(sA[cur], 32, l, hi);           // rows 0-15
    const v16bf a1 = load_a_frag(sA[cur] + 16 * 32, 32, l, hi); // rows 16-31
    __builtin_prefetch(W + (size_t)n0 * K + k0 + 64, 0, 2);
#pragma unroll
    for (int j = 0; j < 4; ++j) {
      const v16bf bfrg =
          load_bT_frag(W + (size_t)(n0 + 16 * j) * K + k0, K, l, hi);
      acc[j]     = wmma_bf16(a0, bfrg, acc[j]);
      acc[j + 4] = wmma_bf16(a1, bfrg, acc[j + 4]);
    }

    if (wv == 0) stage_a_wait();
    __syncthreads();
  }

#pragma unroll
  for (int half = 0; half < 2; ++half) {
#pragma unroll
    for (int r = 0; r < 8; ++r) {
      const int row = m0 + half * 16 + hi * 8 + r;
#pragma unroll
      for (int j = 0; j < 4; ++j) {
        const int col = n0 + 16 * j + l;
        const size_t idx = (size_t)row * N + col;
        const float v = acc[half * 4 + j][r];
        if (MODE == 0) {
          outf[idx] = resid[idx] + v;
        } else if (MODE == 1) {
          const float hv = v + bias[col];
          const float ge = 0.5f * hv * (1.f + erff(hv * 0.70710678118654752f));
          outb[idx] = (bf16)ge;
        } else {
          outf[idx] = resid[idx] + bias[col] + v;
        }
      }
    }
  }
}

// ---------------------------------------------------------------------------
// Host orchestration
// ---------------------------------------------------------------------------
extern "C" void kernel_launch(void* const* d_in, const int* in_sizes, int n_in,
                              void* d_out, int out_size, void* d_ws,
                              size_t ws_size, hipStream_t stream) {
  (void)in_sizes; (void)n_in; (void)out_size; (void)ws_size;
  const float* x         = (const float*)d_in[0];
  const float* alibi     = (const float*)d_in[1];
  const float* rot       = (const float*)d_in[2];
  const float* g_att     = (const float*)d_in[3];
  const float* g_mlp     = (const float*)d_in[4];
  const float* w_qkv     = (const float*)d_in[5];
  const float* w_att_out = (const float*)d_in[6];
  const float* w_mlp_in  = (const float*)d_in[7];
  const float* b_mlp_in  = (const float*)d_in[8];
  const float* w_mlp_out = (const float*)d_in[9];
  const float* b_mlp_out = (const float*)d_in[10];
  float* out = (float*)d_out;

  char* ws = (char*)d_ws;
  const size_t MB = 1u << 20;
  bf16*  yn1   = (bf16*)(ws + 0 * MB);    //  8 MB  [BT,C]
  bf16*  qbuf  = (bf16*)(ws + 8 * MB);    //  8 MB  [B*H,T,D]
  bf16*  kbuf  = (bf16*)(ws + 16 * MB);   //  8 MB  [B*H,T,D]
  bf16*  vTbuf = (bf16*)(ws + 24 * MB);   //  8 MB  [B*H,D,T]
  bf16*  attY  = (bf16*)(ws + 32 * MB);   //  8 MB  [BT,HD]
  float* r1    = (float*)(ws + 40 * MB);  // 16 MB  [BT,C]
  bf16*  yn2   = (bf16*)(ws + 56 * MB);   //  8 MB  [BT,C]
  bf16*  hid   = (bf16*)(ws + 64 * MB);   // 32 MB  [BT,F]
  bf16*  wqkvb = (bf16*)(ws + 96 * MB);   //  6 MB
  bf16*  wattb = (bf16*)(ws + 102 * MB);  //  2 MB
  bf16*  wminb = (bf16*)(ws + 104 * MB);  //  8 MB
  bf16*  wmoutb= (bf16*)(ws + 112 * MB);  //  8 MB  (total 120 MB)

  // one-time weight conversions (L2-resident afterwards)
  f2bf_kernel<<<dim3((3 * HD * Csz + 255) / 256), dim3(256), 0, stream>>>(
      w_qkv, wqkvb, 3 * HD * Csz);
  f2bf_kernel<<<dim3((Csz * HD + 255) / 256), dim3(256), 0, stream>>>(
      w_att_out, wattb, Csz * HD);
  f2bf_kernel<<<dim3((Fsz * Csz + 255) / 256), dim3(256), 0, stream>>>(
      w_mlp_in, wminb, Fsz * Csz);
  f2bf_kernel<<<dim3((Csz * Fsz + 255) / 256), dim3(256), 0, stream>>>(
      w_mlp_out, wmoutb, Csz * Fsz);

  // 1) rmsnorm(x) -> bf16
  rmsnorm_kernel<<<dim3(BT), dim3(256), 0, stream>>>(x, g_att, yn1, Csz);

  // 2) QKV + rope: 256 m-tiles * 48 e-tiles = 12288 waves / 8 per block
  qkv_rope_kernel<<<dim3(1536), dim3(256), 0, stream>>>(yn1, wqkvb, rot, qbuf,
                                                        kbuf, vTbuf);

  // 3) flash attention: 4096 q-tiles, 4 waves per block
  attn_kernel<<<dim3(1024), dim3(128), 0, stream>>>(qbuf, kbuf, vTbuf, alibi,
                                                    attY);

  // 4) out-proj + residual: M=4096 N=1024 K=1024 -> 128x2 block tiles
  gemm32x512_kernel<0><<<dim3(256), dim3(256), 0, stream>>>(
      attY, wattb, nullptr, x, r1, nullptr, BT, Csz, HD);

  // 5) rmsnorm(r1) -> bf16
  rmsnorm_kernel<<<dim3(BT), dim3(256), 0, stream>>>(r1, g_mlp, yn2, Csz);

  // 6) MLP in + gelu: M=4096 N=4096 K=1024 -> 128x8 block tiles
  gemm32x512_kernel<1><<<dim3(1024), dim3(256), 0, stream>>>(
      yn2, wminb, b_mlp_in, nullptr, nullptr, hid, BT, Fsz, Csz);

  // 7) MLP out + bias + residual: M=4096 N=1024 K=4096 -> 128x2 block tiles
  gemm32x512_kernel<2><<<dim3(256), dim3(256), 0, stream>>>(
      hid, wmoutb, b_mlp_out, r1, out, nullptr, BT, Csz, Fsz);
}
